// EfficientGATLayer_70007966925392
// MI455X (gfx1250) — matrix operand order repaired and checked by reading, not compiled, not run on hardware
//
#include <hip/hip_runtime.h>
#include <hip/hip_bf16.h>
#include <stdint.h>

// ---------------------------------------------------------------------------
// GAT layer for MI455X (gfx1250, wave32, WMMA).
//   h = x @ W.T                      -> bf16-split WMMA GEMM (3-term, ~fp32)
//   attn = lrelu(sum(cat(h_s,h_d)*a))-> tiny per-edge wave kernel (E=500)
//   out[dst] += 0.1*attn*h[src]      -> staged scatter with f32 atomics
// ---------------------------------------------------------------------------

#define HEADS     4
#define HEAD_DIM  32
#define IN_F      128
#define OUT_F     128   // HEADS * HEAD_DIM

typedef __attribute__((ext_vector_type(16))) __bf16 v16bf;
typedef __attribute__((ext_vector_type(8)))  float  v8f;

union BFrag {
    uint4 u4[2];   // two ds_load_b128 worth
    v16bf v;       // 16 bf16 = 8 VGPRs, one WMMA B (or A) fragment
};

// ---------------------------------------------------------------------------
// Kernel 1: h = x @ W.T  via v_wmma_f32_16x16x32_bf16 with fp32->bf16 split.
// Block: 256 threads (8 waves). Each block: 128 rows. Each wave: 16x128 tile.
// LDS: W split into bf16 hi/lo planes, padded row stride for bank spread.
// ---------------------------------------------------------------------------
__global__ __launch_bounds__(256) void gat_gemm_wmma(
    const float* __restrict__ x, const float* __restrict__ W,
    float* __restrict__ out, int64_t N)
{
    __shared__ __bf16 sWhi[OUT_F][IN_F + 8];  // +8 bf16 pad -> 272B stride (16B aligned)
    __shared__ __bf16 sWlo[OUT_F][IN_F + 8];

    const int tid  = threadIdx.x;
    const int lane = tid & 31;
    const int wave = tid >> 5;
    const int half = lane >> 4;   // 0: lanes 0-15, 1: lanes 16-31
    const int m    = lane & 15;
    const int64_t blockBase = (int64_t)blockIdx.x * 128;

    // Warm the x tile for this block (global_prefetch_b8); rows are 512B = 4 lines.
    if (tid < 128) {
        int64_t r = blockBase + tid; if (r >= N) r = N - 1;
        const float* pr = x + r * IN_F;
        #pragma unroll
        for (int j = 0; j < 4; ++j) __builtin_prefetch(pr + j * 32, 0, 0);
    }

    // Stage W into LDS as bf16 hi + residual lo (error-split for ~fp32 accuracy).
    // W is row-major [n][k] == exactly B[k][n] fragment source layout.
    for (int i = tid; i < OUT_F * IN_F; i += 256) {
        const int n = i >> 7, k = i & (IN_F - 1);
        const float w  = W[i];
        const __bf16 hi = (__bf16)w;
        const __bf16 lo = (__bf16)(w - (float)hi);
        sWhi[n][k] = hi;
        sWlo[n][k] = lo;
    }
    __syncthreads();

    const int64_t rowBase = blockBase + (int64_t)wave * 16;
    int64_t arow = rowBase + m; if (arow >= N) arow = N - 1;   // clamp (stores guarded)
    const float* __restrict__ xrow = x + arow * IN_F;

    v8f acc[8];
    #pragma unroll
    for (int t = 0; t < 8; ++t) acc[t] = (v8f)0.0f;

    // K = 128 in 4 steps of 32. Per step: build split A fragment, then 8 column
    // tiles x 3 WMMAs (hi*hi + lo*hi + hi*lo). 96 WMMA per wave total.
    #pragma unroll
    for (int ks = 0; ks < 4; ++ks) {
        // A 16x32 fragment, row m = lane%16, contiguous k order (same as B):
        //   elems 0..7  <-> K = half*16 + 0..7
        //   elems 8..15 <-> K = half*16 + 8 + 0..7
        const int k0 = ks * 32 + half * 16;
        v16bf a_hi, a_lo;
        #pragma unroll
        for (int i = 0; i < 8; ++i) {
            const float f0 = xrow[k0 + i];
            const __bf16 h0 = (__bf16)f0;
            a_hi[i] = h0;
            a_lo[i] = (__bf16)(f0 - (float)h0);
            const float f1 = xrow[k0 + 8 + i];
            const __bf16 h1 = (__bf16)f1;
            a_hi[8 + i] = h1;
            a_lo[8 + i] = (__bf16)(f1 - (float)h1);
        }

        // B 32x16 layout: lane = column n, lower/upper half-wave = K 0-15 / 16-31.
        const int kk = ks * 32 + half * 16;
        #pragma unroll
        for (int t = 0; t < 8; ++t) {
            const int col = t * 16 + m;
            BFrag bh, bl;
            const uint4* ph = (const uint4*)&sWhi[col][kk];
            const uint4* pl = (const uint4*)&sWlo[col][kk];
            bh.u4[0] = ph[0]; bh.u4[1] = ph[1];     // ds_load_b128 x2
            bl.u4[0] = pl[0]; bl.u4[1] = pl[1];

            acc[t] = __builtin_amdgcn_wmma_f32_16x16x32_bf16(
                false, a_hi, false, bh.v, (short)0, acc[t], false, false);
            acc[t] = __builtin_amdgcn_wmma_f32_16x16x32_bf16(
                false, a_lo, false, bh.v, (short)0, acc[t], false, false);
            acc[t] = __builtin_amdgcn_wmma_f32_16x16x32_bf16(
                false, a_hi, false, bl.v, (short)0, acc[t], false, false);
        }
    }

    // f32 C/D layout: VGPR r <-> row M = r (lanes 0-15) or 8+r (lanes 16-31),
    // lane%16 = column within tile.
    #pragma unroll
    for (int r = 0; r < 8; ++r) {
        const int64_t orow = rowBase + half * 8 + r;
        if (orow < N) {
            float* op = out + orow * OUT_F + m;
            #pragma unroll
            for (int t = 0; t < 8; ++t) op[t * 16] = acc[t][r];
        }
    }
}

// ---------------------------------------------------------------------------
// Kernel 2: per-(edge,head) attention. One wave32 per (e,h); lane = d.
// Stages attn and gathered h_src into workspace so the scatter phase never
// reads rows that concurrent atomics are mutating.
// ---------------------------------------------------------------------------
__global__ __launch_bounds__(256) void gat_edge_attn(
    const float* __restrict__ h, const int* __restrict__ ei,
    const float* __restrict__ a, float* __restrict__ attn_ws,
    float* __restrict__ hsrc_ws, int E, int64_t N)
{
    const int gid  = blockIdx.x * blockDim.x + threadIdx.x;
    const int w    = gid >> 5;
    const int lane = gid & 31;
    if (w >= E * HEADS) return;
    const int e  = w >> 2;             // / HEADS
    const int hd = w & (HEADS - 1);

    int64_t src = ei[e];
    int64_t dst = ei[E + e];
    if (src < 0) src = 0; if (src >= N) src = N - 1;
    if (dst < 0) dst = 0; if (dst >= N) dst = N - 1;

    const float hs  = h[src * OUT_F + hd * HEAD_DIM + lane];
    const float hdv = h[dst * OUT_F + hd * HEAD_DIM + lane];
    float p = hs  * a[hd * 2 * HEAD_DIM + lane]
            + hdv * a[hd * 2 * HEAD_DIM + HEAD_DIM + lane];

    #pragma unroll
    for (int off = 16; off > 0; off >>= 1) p += __shfl_xor(p, off, 32);

    const float attn = (p > 0.0f) ? p : 0.2f * p;   // leaky_relu(0.2)
    if (lane == 0) attn_ws[w] = attn;
    hsrc_ws[(int64_t)w * HEAD_DIM + lane] = hs;
}

// ---------------------------------------------------------------------------
// Kernel 3: out[dst, h, :] += 0.1 * attn * h_src  (f32 global atomics,
// handles duplicate dst like jnp .at[].add).
// ---------------------------------------------------------------------------
__global__ __launch_bounds__(256) void gat_edge_scatter(
    const float* __restrict__ attn_ws, const float* __restrict__ hsrc_ws,
    const int* __restrict__ ei, float* __restrict__ out, int E, int64_t N)
{
    const int gid  = blockIdx.x * blockDim.x + threadIdx.x;
    const int w    = gid >> 5;
    const int lane = gid & 31;
    if (w >= E * HEADS) return;
    const int e  = w >> 2;
    const int hd = w & (HEADS - 1);

    int64_t dst = ei[E + e];
    if (dst < 0) dst = 0; if (dst >= N) dst = N - 1;

    const float val = 0.1f * attn_ws[w] * hsrc_ws[(int64_t)w * HEAD_DIM + lane];
    atomicAdd(&out[dst * OUT_F + hd * HEAD_DIM + lane], val);
}

// ---------------------------------------------------------------------------
// Launch. Inputs (setup_inputs order): x[N,128] f32, edge_index[2,E] int,
// W[128,128] f32, a[4,64] f32. Output: [N,128] f32.
// ---------------------------------------------------------------------------
extern "C" void kernel_launch(void* const* d_in, const int* in_sizes, int n_in,
                              void* d_out, int out_size, void* d_ws, size_t ws_size,
                              hipStream_t stream)
{
    const float* x  = (const float*)d_in[0];
    const int*   ei = (const int*)d_in[1];
    const float* W  = (const float*)d_in[2];
    const float* a  = (const float*)d_in[3];
    float* out = (float*)d_out;

    const int64_t N = (int64_t)in_sizes[0] / IN_F;   // 200000
    const int     E = in_sizes[1] / 2;               // 500

    // GEMM: writes out = h everywhere (reference out starts as h).
    dim3 g1((unsigned)((N + 127) / 128));
    gat_gemm_wmma<<<g1, 256, 0, stream>>>(x, W, out, N);

    // Edge phase: E*HEADS waves.
    const int nw = E * HEADS;
    float* attn_ws = (float*)d_ws;          // nw floats
    float* hsrc_ws = attn_ws + nw;          // nw * HEAD_DIM floats
    const int blocks = (nw * 32 + 255) / 256;
    if (blocks > 0) {
        gat_edge_attn<<<blocks, 256, 0, stream>>>(out, ei, a, attn_ws, hsrc_ws, E, N);
        gat_edge_scatter<<<blocks, 256, 0, stream>>>(attn_ws, hsrc_ws, ei, out, E, N);
    }
}
